// CausalSelfAttention_72258529788584
// MI455X (gfx1250) — matrix operand order repaired and checked by reading, not compiled, not run on hardware
//
#include <hip/hip_runtime.h>
#include <hip/hip_bf16.h>

// Causal self-attention for MI455X (gfx1250).
// v_wmma_f32_16x16x32_f16 everywhere; double-buffered LDS; attention K/V chunks
// streamed with global_load_async_to_lds_b128 (2 chunks in flight, ASYNCcnt waits).
// B=2, T=2048, C=1024, H=16, D=64.

typedef __attribute__((ext_vector_type(16))) _Float16 v16h;
typedef __attribute__((ext_vector_type(8)))  _Float16 v8h;
typedef __attribute__((ext_vector_type(4)))  _Float16 hv4;
typedef __attribute__((ext_vector_type(8)))  float    v8f;
typedef __attribute__((ext_vector_type(4)))  float    fv4;

__device__ __forceinline__ v8f wmma16x16x32(v16h a, v16h b, v8f c) {
  return __builtin_amdgcn_wmma_f32_16x16x32_f16(false, a, false, b, (short)0, c, false, false);
}

// Async 16B global -> LDS copy (ASYNCcnt). LDS offset = low 32 bits of generic ptr.
__device__ __forceinline__ void async_ld_b128(void* lds_dst, const void* gsrc) {
  unsigned loff = (unsigned)(unsigned long long)lds_dst;
  asm volatile("global_load_async_to_lds_b128 %0, %1, off"
               :: "v"(loff), "v"(gsrc) : "memory");
}
__device__ __forceinline__ void wait_async0() {
  asm volatile("s_wait_asynccnt 0x0" ::: "memory");
}
__device__ __forceinline__ void wait_async2() {   // <=2 outstanding: prev chunk landed
  asm volatile("s_wait_asynccnt 0x2" ::: "memory");
}

#define TDIM   2048
#define CEMB   1024
#define NHEAD  16
#define HDIM   64
#define MROWS  4096
#define NQKV   3072
#define LDPAD  40            // LDS stride in halfs (20 dwords -> conflict-free)
#define KSTR   72            // attention K-tile stride in halfs (36 dwords)

__device__ __forceinline__ v16h cat16(v8h lo, v8h hi) {
  return __builtin_shufflevector(lo, hi, 0,1,2,3,4,5,6,7,8,9,10,11,12,13,14,15);
}

// ---------------------------------------------------------------------------
// Kernel 1: qkv = x @ W_qkv + b_qkv ; scatter Q[b,h,t,d], K[b,h,t,d], V^T[b,h,d,t].
// 128x64 tile per block; 8 waves x (16 rows x 64 cols); double-buffered LDS.
// ---------------------------------------------------------------------------
__global__ __launch_bounds__(256) void qkv_kernel(
    const float* __restrict__ x, const float* __restrict__ Wqkv,
    const float* __restrict__ bqkv,
    _Float16* __restrict__ Qg, _Float16* __restrict__ Kg, _Float16* __restrict__ Vt)
{
  __shared__ __align__(16) _Float16 la[2][128 * LDPAD];   // [row][k]
  __shared__ __align__(16) _Float16 lb[2][64  * LDPAD];   // [n][k]

  const int tid  = threadIdx.x;
  const int lane = tid & 31, wave = tid >> 5;
  const int row  = lane & 15, hlf = lane >> 4;
  const int m0 = blockIdx.y * 128;
  const int n0 = blockIdx.x * 64;

  auto loadAB = [&](int kq, fv4* av, fv4* bv) {
    #pragma unroll
    for (int i = 0; i < 4; ++i) {               // A: 128x32 f32
      int e = tid + i * 256, r = e >> 3, c = (e & 7) * 4;
      av[i] = *(const fv4*)&x[(size_t)(m0 + r) * CEMB + kq + c];
    }
    #pragma unroll
    for (int i = 0; i < 2; ++i) {               // B: 32x64 f32
      int e = tid + i * 256, kk = e >> 4, n4 = (e & 15) * 4;
      bv[i] = *(const fv4*)&Wqkv[(size_t)(kq + kk) * NQKV + n0 + n4];
    }
  };
  auto storeAB = [&](int buf, const fv4* av, const fv4* bv) {
    #pragma unroll
    for (int i = 0; i < 4; ++i) {
      int e = tid + i * 256, r = e >> 3, c = (e & 7) * 4;
      hv4 h; h[0]=(_Float16)av[i][0]; h[1]=(_Float16)av[i][1];
             h[2]=(_Float16)av[i][2]; h[3]=(_Float16)av[i][3];
      *(hv4*)&la[buf][r * LDPAD + c] = h;
    }
    #pragma unroll
    for (int i = 0; i < 2; ++i) {               // transpose scatter [n][k]
      int e = tid + i * 256, kk = e >> 4, n4 = (e & 15) * 4;
      #pragma unroll
      for (int j = 0; j < 4; ++j)
        lb[buf][(n4 + j) * LDPAD + kk] = (_Float16)bv[i][j];
    }
  };

  v8f acc[4] = {};
  fv4 av[4], bv[2];
  loadAB(0, av, bv);
  storeAB(0, av, bv);
  __syncthreads();

  for (int k0 = 0; k0 < CEMB; k0 += 32) {
    const int cur = (k0 >> 5) & 1;
    const bool more = (k0 + 32) < CEMB;
    if (more) loadAB(k0 + 32, av, bv);          // next-tile loads overlap compute

    const _Float16* ar = &la[cur][(wave * 16 + row) * LDPAD];
    v16h af = cat16(*(const v8h*)&ar[hlf * 16], *(const v8h*)&ar[hlf * 16 + 8]);
    v16h bf[4];
    #pragma unroll
    for (int dt = 0; dt < 4; ++dt)
      bf[dt] = *(const v16h*)&lb[cur][(dt * 16 + row) * LDPAD + hlf * 16];
    #pragma unroll
    for (int dt = 0; dt < 4; ++dt)
      acc[dt] = wmma16x16x32(af, bf[dt], acc[dt]);

    if (more) storeAB(cur ^ 1, av, bv);
    __syncthreads();
  }

  // branchless scatter: output select is block-uniform (64-col tile within Q|K|V)
  const int whichu = n0 >> 10;
  _Float16* basep = (whichu == 0) ? Qg : (whichu == 1) ? Kg : Vt;
  const int st_t = (whichu == 2) ? 1 : HDIM;
  const int st_d = (whichu == 2) ? TDIM : 1;
  #pragma unroll
  for (int dt = 0; dt < 4; ++dt) {
    const int n = n0 + dt * 16 + row;
    const float bias = bqkv[n];
    const int c = n & (CEMB - 1), h = c >> 6, d = c & 63;
    #pragma unroll
    for (int v = 0; v < 8; ++v) {
      int m = m0 + wave * 16 + v + 8 * hlf;
      int t = m & (TDIM - 1), bb = m >> 11;
      size_t off = (size_t)(bb * NHEAD + h) * (TDIM * HDIM)
                 + (size_t)t * st_t + (size_t)d * st_d;
      basep[off] = (_Float16)(acc[dt][v] + bias);
    }
  }
}

// ---------------------------------------------------------------------------
// Kernel 2: flash-style causal attention. K/V chunks async-staged into
// double-buffered LDS, two chunks in flight (ASYNCcnt pipeline).
// ---------------------------------------------------------------------------
__global__ __launch_bounds__(256) void attn_kernel(
    const _Float16* __restrict__ Qg, const _Float16* __restrict__ Kg,
    const _Float16* __restrict__ Vt, _Float16* __restrict__ Yg)
{
  __shared__ __align__(16) _Float16 kt[2][32 * KSTR];        // K chunk [t][d]
  __shared__ __align__(16) _Float16 vt[2][64 * LDPAD];       // V chunk [d][t]
  __shared__ __align__(16) _Float16 pbuf[8][16][LDPAD];      // per-wave P transpose

  const int tid  = threadIdx.x;
  const int lane = tid & 31, wave = tid >> 5;
  const int row  = lane & 15, hlf = lane >> 4;
  const int bh = blockIdx.y;
  const int q0 = blockIdx.x * 128;
  const int qw = q0 + wave * 16;

  const _Float16* Qb = Qg + (size_t)bh * TDIM * HDIM;
  const _Float16* Kb = Kg + (size_t)bh * TDIM * HDIM;
  const _Float16* Vb = Vt + (size_t)bh * HDIM * TDIM;

  // Q A-fragments (registers)
  v16h aq[2];
  #pragma unroll
  for (int dc = 0; dc < 2; ++dc) {
    const _Float16* qr = &Qb[(size_t)(qw + row) * HDIM + dc * 32];
    aq[dc] = cat16(*(const v8h*)&qr[hlf * 16], *(const v8h*)&qr[hlf * 16 + 8]);
  }

  v8f accy[4] = {};
  float mrow[8], lrow[8];
  #pragma unroll
  for (int v = 0; v < 8; ++v) { mrow[v] = -1e30f; lrow[v] = 0.0f; }

  const float scale = 0.125f;
  const int kend = qw + 16;          // wave-uniform causal bound
  const int kmax = q0 + 128;         // block-uniform loop bound (>=128)

  const int kr = tid >> 3, kch = tid & 7;   // K staging: 32 rows x 8 x16B
  const int vd = tid >> 2, vch = tid & 3;   // V staging: 64 rows x 4 x16B

  auto stage = [&](int kc, int buf) {
    async_ld_b128(&kt[buf][kr * KSTR + kch * 8], &Kb[(size_t)(kc + kr) * HDIM + kch * 8]);
    async_ld_b128(&vt[buf][vd * LDPAD + vch * 8], &Vb[(size_t)vd * TDIM + kc + vch * 8]);
  };
  stage(0, 0);
  stage(32, 1);

  for (int kc = 0; kc < kmax; kc += 32) {
    const int cur = (kc >> 5) & 1;
    if ((kc + 32) < kmax) wait_async2(); else wait_async0();  // chunk kc landed
    __syncthreads();                                          // ... for every wave

    if (kc < kend) {                                          // wave-uniform branch
      // S = Q K^T for 32 key columns: batch all K fragments, then WMMA chain
      v16h bk0a = *(const v16h*)&kt[cur][ row       * KSTR + hlf * 16];
      v16h bk1a = *(const v16h*)&kt[cur][(row + 16) * KSTR + hlf * 16];
      v16h bk0b = *(const v16h*)&kt[cur][ row       * KSTR + 32 + hlf * 16];
      v16h bk1b = *(const v16h*)&kt[cur][(row + 16) * KSTR + 32 + hlf * 16];
      v8f s0 = {}, s1 = {};
      s0 = wmma16x16x32(aq[0], bk0a, s0);
      s1 = wmma16x16x32(aq[0], bk1a, s1);
      s0 = wmma16x16x32(aq[1], bk0b, s0);
      s1 = wmma16x16x32(aq[1], bk1b, s1);

      // online softmax (row r = qw + v + 8*hlf; cols across 16-lane group)
      float alpha[8];
      #pragma unroll
      for (int v = 0; v < 8; ++v) {
        int r = qw + v + 8 * hlf;
        float x0 = s0[v] * scale, x1 = s1[v] * scale;
        if (kc + row > r)      x0 = -1e30f;
        if (kc + 16 + row > r) x1 = -1e30f;
        float bm = fmaxf(x0, x1);
        bm = fmaxf(bm, __shfl_xor(bm, 1));
        bm = fmaxf(bm, __shfl_xor(bm, 2));
        bm = fmaxf(bm, __shfl_xor(bm, 4));
        bm = fmaxf(bm, __shfl_xor(bm, 8));
        float mn = fmaxf(mrow[v], bm);
        alpha[v] = __expf(mrow[v] - mn);
        mrow[v] = mn;
        float e0 = __expf(x0 - mn);
        float e1 = __expf(x1 - mn);
        float rs = e0 + e1;
        rs += __shfl_xor(rs, 1);
        rs += __shfl_xor(rs, 2);
        rs += __shfl_xor(rs, 4);
        rs += __shfl_xor(rs, 8);
        lrow[v] = lrow[v] * alpha[v] + rs;
        pbuf[wave][v + 8 * hlf][row]      = (_Float16)e0;
        pbuf[wave][v + 8 * hlf][row + 16] = (_Float16)e1;
      }
      #pragma unroll
      for (int dt = 0; dt < 4; ++dt)
        #pragma unroll
        for (int v = 0; v < 8; ++v)
          accy[dt][v] *= alpha[v];

      // P as A fragment (per-wave LDS, dscnt-ordered); Y += P @ V
      const _Float16* pr = &pbuf[wave][row][0];
      v16h ap = cat16(*(const v8h*)&pr[hlf * 16], *(const v8h*)&pr[hlf * 16 + 8]);
      v16h bvv[4];
      #pragma unroll
      for (int dt = 0; dt < 4; ++dt)
        bvv[dt] = *(const v16h*)&vt[cur][(dt * 16 + row) * LDPAD + hlf * 16];
      #pragma unroll
      for (int dt = 0; dt < 4; ++dt)
        accy[dt] = wmma16x16x32(ap, bvv[dt], accy[dt]);
    }

    __syncthreads();                         // all readers done with buf `cur`
    if (kc + 64 < kmax) stage(kc + 64, cur); // refill freed buffer
  }

  float rl[8];
  #pragma unroll
  for (int v = 0; v < 8; ++v) rl[v] = 1.0f / lrow[v];
  const int b = bh >> 4, h = bh & 15;
  #pragma unroll
  for (int dt = 0; dt < 4; ++dt) {
    #pragma unroll
    for (int v = 0; v < 8; ++v) {
      int t = qw + v + 8 * hlf;
      int d = h * HDIM + dt * 16 + row;
      Yg[((size_t)(b * TDIM + t)) * CEMB + d] = (_Float16)(accy[dt][v] * rl[v]);
    }
  }
}

// ---------------------------------------------------------------------------
// Kernel 3: out = Y @ W_proj + b_proj (f32 out); double-buffered LDS,
// A tile is f16 pass-through, B tile f32->f16 transposed.
// ---------------------------------------------------------------------------
__global__ __launch_bounds__(256) void proj_kernel(
    const _Float16* __restrict__ Yg, const float* __restrict__ Wp,
    const float* __restrict__ bp, float* __restrict__ out)
{
  __shared__ __align__(16) _Float16 la[2][128 * LDPAD];
  __shared__ __align__(16) _Float16 lb[2][64  * LDPAD];

  const int tid  = threadIdx.x;
  const int lane = tid & 31, wave = tid >> 5;
  const int row  = lane & 15, hlf = lane >> 4;
  const int m0 = blockIdx.y * 128;
  const int n0 = blockIdx.x * 64;

  const int ar_ = tid >> 1, ac_ = (tid & 1) * 16;   // A: 128 rows x 2 x16-half chunks

  auto loadAB = [&](int kq, v16h* ah, fv4* bv) {
    ah[0] = *(const v16h*)&Yg[(size_t)(m0 + ar_) * CEMB + kq + ac_];
    #pragma unroll
    for (int i = 0; i < 2; ++i) {
      int e = tid + i * 256, kk = e >> 4, n4 = (e & 15) * 4;
      bv[i] = *(const fv4*)&Wp[(size_t)(kq + kk) * CEMB + n0 + n4];
    }
  };
  auto storeAB = [&](int buf, const v16h* ah, const fv4* bv) {
    *(v16h*)&la[buf][ar_ * LDPAD + ac_] = ah[0];
    #pragma unroll
    for (int i = 0; i < 2; ++i) {
      int e = tid + i * 256, kk = e >> 4, n4 = (e & 15) * 4;
      #pragma unroll
      for (int j = 0; j < 4; ++j)
        lb[buf][(n4 + j) * LDPAD + kk] = (_Float16)bv[i][j];
    }
  };

  v8f acc[4] = {};
  v16h ah[1]; fv4 bv[2];
  loadAB(0, ah, bv);
  storeAB(0, ah, bv);
  __syncthreads();

  for (int k0 = 0; k0 < CEMB; k0 += 32) {
    const int cur = (k0 >> 5) & 1;
    const bool more = (k0 + 32) < CEMB;
    if (more) loadAB(k0 + 32, ah, bv);

    const _Float16* ar = &la[cur][(wave * 16 + row) * LDPAD];
    v16h af = cat16(*(const v8h*)&ar[hlf * 16], *(const v8h*)&ar[hlf * 16 + 8]);
    v16h bf[4];
    #pragma unroll
    for (int dt = 0; dt < 4; ++dt)
      bf[dt] = *(const v16h*)&lb[cur][(dt * 16 + row) * LDPAD + hlf * 16];
    #pragma unroll
    for (int dt = 0; dt < 4; ++dt)
      acc[dt] = wmma16x16x32(af, bf[dt], acc[dt]);

    if (more) storeAB(cur ^ 1, ah, bv);
    __syncthreads();
  }

  #pragma unroll
  for (int dt = 0; dt < 4; ++dt) {
    const int n = n0 + dt * 16 + row;
    const float bias = bp[n];
    #pragma unroll
    for (int v = 0; v < 8; ++v) {
      int m = m0 + wave * 16 + v + 8 * hlf;
      out[(size_t)m * CEMB + n] = acc[dt][v] + bias;
    }
  }
}

// ---------------------------------------------------------------------------
extern "C" void kernel_launch(void* const* d_in, const int* in_sizes, int n_in,
                              void* d_out, int out_size, void* d_ws, size_t ws_size,
                              hipStream_t stream) {
  const float* x    = (const float*)d_in[0];
  const float* Wqkv = (const float*)d_in[1];
  const float* bqkv = (const float*)d_in[2];
  const float* Wp   = (const float*)d_in[3];
  const float* bp   = (const float*)d_in[4];
  float* out = (float*)d_out;

  const size_t SEG = (size_t)32 * TDIM * HDIM;   // 4,194,304 halfs per tensor
  _Float16* Qg = (_Float16*)d_ws;
  _Float16* Kg = Qg + SEG;
  _Float16* Vt = Kg + SEG;
  _Float16* Yg = Vt + SEG;

  qkv_kernel<<<dim3(NQKV / 64, MROWS / 128), 256, 0, stream>>>(x, Wqkv, bqkv, Qg, Kg, Vt);
  attn_kernel<<<dim3(TDIM / 128, 2 * NHEAD), 256, 0, stream>>>(Qg, Kg, Vt, Yg);
  proj_kernel<<<dim3(CEMB / 64, MROWS / 128), 256, 0, stream>>>(Yg, Wp, bp, out);
}
